// EdgeConvBlock_70334384439906
// MI455X (gfx1250) — matrix operand order repaired and checked by reading, not compiled, not run on hardware
//
#include <hip/hip_runtime.h>
#include <hip/hip_bf16.h>

// ---------------------------------------------------------------------------
// EdgeConv block for MI455X (gfx1250, wave32, WMMA + TDM).
//   B=64, N=1024, C_IN=64, K=16, all layer widths 64.
//   k_prep  : features (B,C,N) f32 -> ft (B,N,C) f16 (point-major for gather)
//   k_wprep : W0|W1|W2 f32 -> contiguous f16 blob (TDM source)
//   k_knn   : per-point top-17 insertion sort over 1024 2-D candidates
//   k_sc    : shortcut GEMM (fp32) + its BN stats
//   k_pass<D>: fused conv stack, weights staged to LDS via tensor_load_to_lds;
//              D<3 accumulates global BN stats for layer D, D==3 writes output.
//   k_fin   : (sum,sumsq) -> per-channel BN scale/shift.
// Working set ~29 MB => resident in 192 MB L2; HBM touch ~34 MB (~1.5us @23.3TB/s).
// ---------------------------------------------------------------------------

typedef _Float16 h16;
typedef __attribute__((ext_vector_type(16))) _Float16 v16h;
typedef __attribute__((ext_vector_type(8)))  _Float16 v8h;
typedef __attribute__((ext_vector_type(8)))  float    v8f;
typedef __attribute__((ext_vector_type(4))) unsigned int u32x4;
typedef __attribute__((ext_vector_type(8))) unsigned int u32x8;

#define NB   64
#define NP   1024
#define CIN  64
#define KNN  16
#define WBLOB_HALVES 16384   // 8192 (W0) + 4096 (W1) + 4096 (W2)
#define WBLOB_DWORDS 8192

__device__ __forceinline__ v8f wmma_f16(v16h a, v16h b, v8f c) {
  // (neg_a, A, neg_b, B, c_mod, C, reuse_a, reuse_b)
  return __builtin_amdgcn_wmma_f32_16x16x32_f16(false, a, false, b, (short)0, c, false, false);
}

__device__ __forceinline__ v16h cat8(v8h a, v8h b) {
  return __builtin_shufflevector(a, b, 0,1,2,3,4,5,6,7,8,9,10,11,12,13,14,15);
}

// A fragment (16x32 M-by-K tile) from LDS weights W[row][CK], K window at kbase.
// Layout (ISA 7.12.2): lane m=lane&15 holds row row0+m; lane-hi selects K sub-bank.
__device__ __forceinline__ v16h afrag_lds(const h16* W, int CK, int row0, int kbase, int lane) {
  int mm = lane & 15, hh = lane >> 4;
  const h16* rp = W + (size_t)(row0 + mm) * CK + kbase + hh * 16;
  v16h a;
#pragma unroll
  for (int j = 0; j < 8; ++j) a[j]     = rp[j];        // K = kbase + hi*16 + j
#pragma unroll
  for (int j = 0; j < 8; ++j) a[j + 8] = rp[8 + j];    // K = kbase + hi*16 + 8 + j
  return a;
}

// B fragment (32x16 K-by-N) from a col-major 16x64 LDS activation tile
// sA[col*64 + c]: contiguous 16 halves -> two ds_load_b128.
__device__ __forceinline__ v16h bfrag_act(const h16* sA, int c0, int col) {
  const v8h* p = (const v8h*)(sA + col * 64 + c0);
  return cat8(p[0], p[1]);
}

// ------------------------------ k_prep ------------------------------------
__global__ __launch_bounds__(256) void k_prep(const float* __restrict__ f, h16* __restrict__ ft) {
  __shared__ float sT[64][65];
  int b  = blockIdx.x >> 4;
  int n0 = (blockIdx.x & 15) << 6;
  int tid = threadIdx.x;
  for (int i = tid; i < 4096; i += 256) {
    int c = i >> 6, n = i & 63;
    sT[c][n] = f[(((size_t)(b << 6) + c) << 10) + n0 + n];       // coalesced read
  }
  __syncthreads();
  for (int i = tid; i < 4096; i += 256) {
    int n = i >> 6, c = i & 63;
    ft[((size_t)((b << 10) + n0 + n) << 6) + c] = (h16)sT[c][n]; // contiguous write
  }
}

// ------------------------------ k_wprep -----------------------------------
__global__ __launch_bounds__(256) void k_wprep(const float* __restrict__ W0, const float* __restrict__ W1,
                                               const float* __restrict__ W2, h16* __restrict__ blob) {
  int i = blockIdx.x * 256 + threadIdx.x;
  if (i < 8192)       blob[i] = (h16)W0[i];
  else if (i < 12288) blob[i] = (h16)W1[i - 8192];
  else if (i < 16384) blob[i] = (h16)W2[i - 12288];
}

// ------------------------------ k_knn -------------------------------------
__global__ __launch_bounds__(256) void k_knn(const float* __restrict__ pts, int* __restrict__ idx) {
  __shared__ float sx[NP], sy[NP];
  int b  = blockIdx.x >> 2;
  int qn = ((blockIdx.x & 3) << 8) + threadIdx.x;
  const float* px = pts + (size_t)b * 2 * NP;
  for (int i = threadIdx.x; i < NP; i += 256) { sx[i] = px[i]; sy[i] = px[NP + i]; }
  __syncthreads();
  float qx = sx[qn], qy = sy[qn];
  float d[17]; int id[17];
#pragma unroll
  for (int j = 0; j < 17; ++j) { d[j] = 3.4e38f; id[j] = 0; }
  for (int m = 0; m < NP; ++m) {
    float dx = sx[m] - qx, dy = sy[m] - qy;
    float dist = dx * dx + dy * dy;
    if (dist < d[16]) {
      // fully predicated sorted-insert (no dynamic indexing -> stays in VGPRs)
#pragma unroll
      for (int j = 16; j >= 1; --j) {
        if      (dist < d[j - 1]) { d[j] = d[j - 1]; id[j] = id[j - 1]; }
        else if (dist < d[j])     { d[j] = dist;     id[j] = m; }
      }
      if (dist < d[0]) { d[0] = dist; id[0] = m; }
    }
  }
  int* op = idx + ((size_t)((b << 10) + qn) << 4);
#pragma unroll
  for (int j = 0; j < 16; ++j) op[j] = id[j + 1];   // drop self
}

// ------------------------------ k_sc --------------------------------------
__global__ __launch_bounds__(256) void k_sc(const float* __restrict__ f, const float* __restrict__ Wsc,
                                            float* __restrict__ sc_lin, float* __restrict__ gstats) {
  __shared__ float sF[64 * 32];
  int tid = threadIdx.x;
  int b  = blockIdx.x >> 5;
  int n0 = (blockIdx.x & 31) << 5;
  for (int i = tid; i < 2048; i += 256) {
    int c = i >> 5, nl = i & 31;
    sF[i] = f[(((size_t)(b << 6) + c) << 10) + n0 + nl];
  }
  __syncthreads();
  int oq = tid >> 5, nl = tid & 31;
#pragma unroll 1
  for (int i = 0; i < 8; ++i) {
    int o = oq * 8 + i;
    float acc = 0.f;
#pragma unroll
    for (int c = 0; c < 64; ++c) acc = fmaf(Wsc[(o << 6) + c], sF[(c << 5) + nl], acc);
    sc_lin[(((size_t)(b << 6) + o) << 10) + n0 + nl] = acc;
    float s = acc, q = acc * acc;
#pragma unroll
    for (int msk = 16; msk >= 1; msk >>= 1) { s += __shfl_xor(s, msk, 32); q += __shfl_xor(q, msk, 32); }
    if (nl == 0) { atomicAdd(&gstats[3 * 128 + o * 2], s); atomicAdd(&gstats[3 * 128 + o * 2 + 1], q); }
  }
}

// ------------------------------ k_fin -------------------------------------
__global__ void k_fin(const float* __restrict__ gstats, const float* __restrict__ g,
                      const float* __restrict__ bta, float* __restrict__ bnp,
                      int layer, float invcnt) {
  int c = threadIdx.x;
  if (c < 64) {
    float s = gstats[layer * 128 + c * 2], q = gstats[layer * 128 + c * 2 + 1];
    float mean = s * invcnt;
    float var  = q * invcnt - mean * mean;
    float sc   = g[c] * rsqrtf(var + 1e-5f);
    bnp[layer * 128 + c * 2]     = sc;
    bnp[layer * 128 + c * 2 + 1] = bta[c] - mean * sc;
  }
}

__global__ void k_zero(float* p, int n) {
  int i = blockIdx.x * 256 + threadIdx.x;
  if (i < n) p[i] = 0.f;
}

// --------------------------- fused conv passes -----------------------------
// BN+ReLU and write the 64x16 activation tile col-major: one ds_store_b128
// per o-tile per lane (8 packed halves).
__device__ __forceinline__ void bn_relu_store(const v8f* acc, const float* bnp,
                                              h16* sA, int col, int hi) {
#pragma unroll
  for (int ot = 0; ot < 4; ++ot) {
    v8h pk;
#pragma unroll
    for (int r = 0; r < 8; ++r) {
      int o = ot * 16 + r + 8 * hi;                     // C/D layout channel
      pk[r] = (h16)fmaxf(bnp[o * 2] * acc[ot][r] + bnp[o * 2 + 1], 0.f);
    }
    *(v8h*)(sA + col * 64 + ot * 16 + 8 * hi) = pk;     // 16B aligned
  }
}

__device__ __forceinline__ void acc_stats(const v8f* acc, float* psum, float* psq) {
#pragma unroll
  for (int ot = 0; ot < 4; ++ot)
#pragma unroll
    for (int r = 0; r < 8; ++r) {
      float e = acc[ot][r];
      psum[ot * 8 + r] += e;
      psq[ot * 8 + r]  += e * e;
    }
}

template <int DEPTH>
__global__ __launch_bounds__(256) void k_pass(const h16* __restrict__ ft, const int* __restrict__ idx,
                                              const h16* __restrict__ wblob, const float* __restrict__ bnp,
                                              float* __restrict__ gstats, const float* __restrict__ sc_lin,
                                              float* __restrict__ out) {
  __shared__ h16  sW[WBLOB_HALVES];    // W0(64x128) | W1(64x64) | W2(64x64), f16
  __shared__ h16  sAct[8][16 * 64];    // per-wave slab, col-major [col*64 + c]
  __shared__ float sBNP[512];
  __shared__ float sRed[128];

  int tid = threadIdx.x;
  int wv = tid >> 5, lane = tid & 31;

  // --- stage all weights with one Tensor Data Mover transfer (32 KB) ---
  if (wv == 0) {
    unsigned long long ga = (unsigned long long)wblob;      // 57-bit global addr
    unsigned lds_off = (unsigned)(unsigned long long)&sW[0];
    u32x4 g0;
    g0[0] = 1u;                                             // count=1, user D#
    g0[1] = lds_off;                                        // lds_addr
    g0[2] = (unsigned)ga;                                   // global_addr[31:0]
    g0[3] = (unsigned)((ga >> 32) & 0x01FFFFFFu) | (2u << 30); // addr[56:32]|type=2
    u32x8 g1;
    g1[0] = (2u << 16);                // wg_mask=0, data_size=2 -> 4B elements
    g1[1] = (unsigned)(WBLOB_DWORDS & 0xFFFF) << 16;        // tensor_dim0 lo16
    g1[2] = (1u << 16) | (unsigned)(WBLOB_DWORDS >> 16);    // dim0 hi16 | tensor_dim1=1
    g1[3] = (unsigned)WBLOB_DWORDS << 16;                   // tile_dim0 = 8192
    g1[4] = 0u;                                             // tile_dim1/2 unused
    g1[5] = (unsigned)WBLOB_DWORDS;                         // tensor_dim0_stride lo
    g1[6] = 0u;
    g1[7] = 0u;
    asm volatile("tensor_load_to_lds %0, %1" :: "s"(g0), "s"(g1) : "memory");
    __builtin_amdgcn_s_wait_tensorcnt(0);
  }
  for (int i = tid; i < 512; i += 256) sBNP[i] = bnp[i];
  if (tid < 128) sRed[tid] = 0.f;
  __syncthreads();

  const h16* sW0c = sW;
  const h16* sW1c = sW + 8192;
  const h16* sW2c = sW + 12288;

  int col = lane & 15, hi = lane >> 4;
  int b  = blockIdx.x >> 4;            // 16 blocks per batch, 64 points per block
  int n0 = (blockIdx.x & 15) << 6;
  h16* sA = sAct[wv];

  float psum[32], psq[32];
#pragma unroll
  for (int i = 0; i < 32; ++i) { psum[i] = 0.f; psq[i] = 0.f; }

  for (int p = 0; p < 8; ++p) {
    int n = n0 + wv * 8 + p;
    int nbr = idx[((size_t)((b << 10) + n) << 4) + col];
    const h16* pc = ft + ((size_t)((b << 10) + n)   << 6);
    const h16* pn = ft + ((size_t)((b << 10) + nbr) << 6);
    if (p < 7) __builtin_prefetch(pc + 64, 0, 1);      // global_prefetch_b8: next point

    // ---- layer 0: out(64x16) = W0(64x128) x Xedge(128x16) ----
    v8f acc0[4] = {};
#pragma unroll
    for (int kk = 0; kk < 4; ++kk) {
      int c0 = kk * 32 + hi * 16;
      v16h bf;
      if (c0 < 64) {                                    // center half (broadcast xc)
        const v8h* pp = (const v8h*)(pc + c0);
        bf = cat8(pp[0], pp[1]);
      } else {                                          // neighbor - center half
        int cc = c0 - 64;
        const v8h* pa = (const v8h*)(pn + cc);
        const v8h* pb = (const v8h*)(pc + cc);
        bf = cat8(pa[0] - pb[0], pa[1] - pb[1]);
      }
#pragma unroll
      for (int ot = 0; ot < 4; ++ot)
        acc0[ot] = wmma_f16(afrag_lds(sW0c, 128, ot * 16, kk * 32, lane), bf, acc0[ot]);
    }

    if constexpr (DEPTH == 0) {
      acc_stats(acc0, psum, psq);
    } else {
      bn_relu_store(acc0, sBNP + 0, sA, col, hi);

      // ---- layer 1 ----
      v8f acc1[4] = {};
#pragma unroll
      for (int kk = 0; kk < 2; ++kk) {
        v16h bf = bfrag_act(sA, kk * 32 + hi * 16, col);
#pragma unroll
        for (int ot = 0; ot < 4; ++ot)
          acc1[ot] = wmma_f16(afrag_lds(sW1c, 64, ot * 16, kk * 32, lane), bf, acc1[ot]);
      }

      if constexpr (DEPTH == 1) {
        acc_stats(acc1, psum, psq);
      } else {
        bn_relu_store(acc1, sBNP + 128, sA, col, hi);

        // ---- layer 2 ----
        v8f acc2[4] = {};
#pragma unroll
        for (int kk = 0; kk < 2; ++kk) {
          v16h bf = bfrag_act(sA, kk * 32 + hi * 16, col);
#pragma unroll
          for (int ot = 0; ot < 4; ++ot)
            acc2[ot] = wmma_f16(afrag_lds(sW2c, 64, ot * 16, kk * 32, lane), bf, acc2[ot]);
        }

        if constexpr (DEPTH == 2) {
          acc_stats(acc2, psum, psq);
        } else {
          // ---- bn2 + relu + mean_k + shortcut + final relu ----
#pragma unroll
          for (int ot = 0; ot < 4; ++ot)
#pragma unroll
            for (int r = 0; r < 8; ++r) {
              int o = ot * 16 + r + 8 * hi;
              float e = fmaxf(sBNP[256 + o * 2] * acc2[ot][r] + sBNP[256 + o * 2 + 1], 0.f);
#pragma unroll
              for (int msk = 1; msk <= 8; msk <<= 1) e += __shfl_xor(e, msk, 32);
              if (col == 0) {
                float fts = e * 0.0625f;                       // mean over 16 neighbors
                size_t oi = (((size_t)(b << 6) + o) << 10) + n;
                float scb = sBNP[384 + o * 2] * sc_lin[oi] + sBNP[384 + o * 2 + 1];
                out[oi] = fmaxf(scb + fts, 0.f);
              }
            }
        }
      }
    }
  }

  if constexpr (DEPTH < 3) {
    // reduce stats across the 16 column-lanes (masks <16 keep lane-hi groups intact)
#pragma unroll
    for (int i = 0; i < 32; ++i) {
      float s = psum[i], q = psq[i];
#pragma unroll
      for (int msk = 1; msk <= 8; msk <<= 1) { s += __shfl_xor(s, msk, 32); q += __shfl_xor(q, msk, 32); }
      if (col == 0) {
        int ot = i >> 3, r = i & 7;
        int o = ot * 16 + r + 8 * hi;
        atomicAdd(&sRed[o * 2], s);
        atomicAdd(&sRed[o * 2 + 1], q);
      }
    }
    __syncthreads();
    for (int i = tid; i < 128; i += 256) atomicAdd(gstats + DEPTH * 128 + i, sRed[i]);
  }
}

// ------------------------------ launch -------------------------------------
extern "C" void kernel_launch(void* const* d_in, const int* in_sizes, int n_in,
                              void* d_out, int out_size, void* d_ws, size_t ws_size,
                              hipStream_t stream) {
  const float* points   = (const float*)d_in[0];
  const float* features = (const float*)d_in[1];
  const float* W0  = (const float*)d_in[2];
  const float* g0  = (const float*)d_in[3];
  const float* b0  = (const float*)d_in[4];
  const float* W1  = (const float*)d_in[5];
  const float* g1  = (const float*)d_in[6];
  const float* b1  = (const float*)d_in[7];
  const float* W2  = (const float*)d_in[8];
  const float* g2  = (const float*)d_in[9];
  const float* b2  = (const float*)d_in[10];
  const float* Wsc = (const float*)d_in[11];
  const float* gsc = (const float*)d_in[12];
  const float* bsc = (const float*)d_in[13];
  float* out = (float*)d_out;

  // workspace layout (~28 MB + 40 KB, all L2-resident)
  char* ws = (char*)d_ws;
  h16*   ft     = (h16*)ws;                                 //  8 MB  (B,N,C) f16
  int*   idx    = (int*)(ws + ((size_t)8 << 20));           //  4 MB  (B,N,16)
  float* sc_lin = (float*)(ws + ((size_t)12 << 20));        // 16 MB  (B,64,N)
  float* gstats = (float*)(ws + ((size_t)28 << 20));        //  2 KB  4 layers x 64ch x {sum,sumsq}
  float* bnp    = (float*)(ws + ((size_t)28 << 20) + 4096); //  2 KB  4 layers x 64ch x {scale,shift}
  h16*   wblob  = (h16*)(ws + ((size_t)28 << 20) + 8192);   // 32 KB  W0|W1|W2 f16 (TDM source)

  const float inv_conv = 1.0f / (float)(NB * NP * KNN);     // BN over (b,n,k)
  const float inv_sc   = 1.0f / (float)(NB * NP);           // BN over (b,n)

  k_zero<<<2, 256, 0, stream>>>(gstats, 512);
  k_prep<<<NB * 16, 256, 0, stream>>>(features, ft);
  k_wprep<<<64, 256, 0, stream>>>(W0, W1, W2, wblob);
  k_knn<<<NB * 4, 256, 0, stream>>>(points, idx);
  k_sc<<<NB * 32, 256, 0, stream>>>(features, Wsc, sc_lin, gstats);
  k_fin<<<1, 64, 0, stream>>>(gstats, gsc, bsc, bnp, 3, inv_sc);

  k_pass<0><<<NB * 16, 256, 0, stream>>>(ft, idx, wblob, bnp, gstats, sc_lin, out);
  k_fin<<<1, 64, 0, stream>>>(gstats, g0, b0, bnp, 0, inv_conv);
  k_pass<1><<<NB * 16, 256, 0, stream>>>(ft, idx, wblob, bnp, gstats, sc_lin, out);
  k_fin<<<1, 64, 0, stream>>>(gstats, g1, b1, bnp, 1, inv_conv);
  k_pass<2><<<NB * 16, 256, 0, stream>>>(ft, idx, wblob, bnp, gstats, sc_lin, out);
  k_fin<<<1, 64, 0, stream>>>(gstats, g2, b2, bnp, 2, inv_conv);
  k_pass<3><<<NB * 16, 256, 0, stream>>>(ft, idx, wblob, bnp, gstats, sc_lin, out);
}
